// MSA2Pair_188978561514
// MI455X (gfx1250) — compile-verified, matrix operand order checked by reading, not run on hardware
//
#include <hip/hip_runtime.h>
#include <hip/hip_bf16.h>

// ---- problem constants (fixed by reference setup_inputs) ----
#define NSEQ 128      // N
#define LPOS 256      // L
#define DM   256      // d_msa
#define DP   128      // d_pair
#define DH   32       // d_h
#define ROWS (NSEQ*LPOS)   // 32768 msa rows
#define LM   (LPOS*LPOS)   // 65536 pair rows
#define PAD  258           // padded spatial dim

typedef __attribute__((ext_vector_type(16))) _Float16     v16h;
typedef __attribute__((ext_vector_type(8)))  float        v8f;
typedef __attribute__((ext_vector_type(4)))  unsigned int v4u;
typedef __attribute__((ext_vector_type(8)))  unsigned int v8u;

union FragCast { v8u u; v16h h; };

__device__ __forceinline__ v8f vzero8() {
  v8f z;
#pragma unroll
  for (int i = 0; i < 8; ++i) z[i] = 0.0f;
  return z;
}

// Load one 16x32 f16 fragment from a row-major, K-contiguous matrix.
// p points at element [row0][k0] (16B aligned, ld multiple of 8 halfs).
// lane t: row = t&15, K-subblock = (t>>4)*8 within each 16-wide K half.
__device__ __forceinline__ v16h load_frag(const _Float16* p, int ld, int lane) {
  const _Float16* q = p + (lane & 15) * ld + ((lane >> 4) << 3);
  v4u lo = *(const v4u*)q;
  v4u hi = *(const v4u*)(q + 16);
  FragCast f;
#pragma unroll
  for (int i = 0; i < 4; ++i) { f.u[i] = lo[i]; f.u[i + 4] = hi[i]; }
  return f.h;
}

__device__ __forceinline__ v8f wmma_f16(v16h a, v16h b, v8f c) {
  return __builtin_amdgcn_wmma_f32_16x16x32_f16(false, a, false, b, (short)0, c,
                                                false, false);
}

// ------------------------------------------------------------------
// init / prep kernels
// ------------------------------------------------------------------
__global__ void k_zero(unsigned int* p, long n) {
  long i = (long)blockIdx.x * blockDim.x + threadIdx.x;
  long stride = (long)gridDim.x * blockDim.x;
  for (; i < n; i += stride) p[i] = 0u;
}

__global__ void k_prep_weights(const float* __restrict__ Wl,
                               const float* __restrict__ Wr,
                               const float* __restrict__ Wo,
                               const float* __restrict__ Wd,
                               const float* __restrict__ c1,
                               const float* __restrict__ c2,
                               _Float16* __restrict__ WlT, _Float16* __restrict__ WrT,
                               _Float16* __restrict__ WoT, _Float16* __restrict__ WdT,
                               _Float16* __restrict__ Wt1, _Float16* __restrict__ Wt2) {
  int idx = blockIdx.x * 256 + threadIdx.x;
  if (idx < DH * DM) {                 // WlT/WrT: [32][256] <- [256][32]
    int n = idx >> 8, k = idx & 255;
    WlT[idx] = (_Float16)Wl[k * DH + n];
    WrT[idx] = (_Float16)Wr[k * DH + n];
  }
  if (idx < DP * (DH * DH)) {          // WoT: [128][1024] <- [1024][128]
    int c = idx >> 10, k = idx & 1023;
    WoT[idx] = (_Float16)Wo[k * DP + c];
  }
  if (idx < DP * (2 * DP)) {           // WdT: [128][256] <- [256][128]
    int c = idx >> 8, k = idx & 255;
    WdT[idx] = (_Float16)Wd[k * DP + c];
  }
  if (idx < 9 * DP * DP) {             // Wt: [tap][co][ci] <- [co][ci][3][3]
    int tap = idx / (DP * DP), rem = idx & (DP * DP - 1);
    int co = rem >> 7, ci = rem & 127;
    Wt1[idx] = (_Float16)c1[(co * DP + ci) * 9 + tap];
    Wt2[idx] = (_Float16)c2[(co * DP + ci) * 9 + tap];
  }
}

__global__ void k_prep_pair(const float* __restrict__ pair,
                            _Float16* __restrict__ cat) {
  long idx = (long)blockIdx.x * 256 + threadIdx.x;   // over LM*DP
  long row = idx >> 7;
  int c = (int)(idx & 127);
  cat[row * 256 + c] = (_Float16)pair[idx];
}

// ------------------------------------------------------------------
// 1) LayerNorm + left/right projections (WMMA), store transposed f16
//    ltT/rtT layout: [l*32+i][s]  (K-contiguous in s for the big GEMM)
// ------------------------------------------------------------------
__global__ __launch_bounds__(128) void k_ln_proj(
    const float* __restrict__ msa, const float* __restrict__ g,
    const float* __restrict__ b, const _Float16* __restrict__ WlT,
    const float* __restrict__ bl, const _Float16* __restrict__ WrT,
    const float* __restrict__ br, _Float16* __restrict__ ltT,
    _Float16* __restrict__ rtT) {
  __shared__ _Float16 tile[16 * DM];
  __shared__ float red[16][8][2];
  __shared__ float mrs[16][2];
  int tid = threadIdx.x;
  int row = tid >> 3, seg = tid & 7;
  long gbase = (long)(blockIdx.x * 16 + row) * DM + seg * 32;
  float v[32];
  const float4* src = (const float4*)(msa + gbase);
  float s = 0.f, s2 = 0.f;
#pragma unroll
  for (int i = 0; i < 8; ++i) {
    float4 t = src[i];
    v[4 * i + 0] = t.x; v[4 * i + 1] = t.y;
    v[4 * i + 2] = t.z; v[4 * i + 3] = t.w;
  }
#pragma unroll
  for (int i = 0; i < 32; ++i) { s += v[i]; s2 += v[i] * v[i]; }
  red[row][seg][0] = s; red[row][seg][1] = s2;
  __syncthreads();
  if (tid < 16) {
    float ts = 0.f, ts2 = 0.f;
#pragma unroll
    for (int i = 0; i < 8; ++i) { ts += red[tid][i][0]; ts2 += red[tid][i][1]; }
    float mean = ts * (1.0f / DM);
    float var = ts2 * (1.0f / DM) - mean * mean;
    mrs[tid][0] = mean;
    mrs[tid][1] = rsqrtf(var + 1e-5f);
  }
  __syncthreads();
  float mean = mrs[row][0], rstd = mrs[row][1];
#pragma unroll
  for (int i = 0; i < 32; ++i) {
    int c = seg * 32 + i;
    float xn = (v[i] - mean) * rstd * g[c] + b[c];
    tile[row * DM + c] = (_Float16)xn;
  }
  __syncthreads();

  int w = tid >> 5, lane = tid & 31;
  int proj = w >> 1, nt = w & 1;     // proj 0=left 1=right, nt = 16-col half
  const _Float16* WT = proj ? WrT : WlT;
  const float* bias = proj ? br : bl;
  v8f acc = vzero8();
#pragma unroll
  for (int kk = 0; kk < 8; ++kk) {
    v16h a = load_frag(tile + kk * 32, DM, lane);
    v16h bf = load_frag(WT + nt * 16 * DM + kk * 32, DM, lane);
    acc = wmma_f16(a, bf, acc);
  }
  int i = nt * 16 + (lane & 15);
  float bv = bias[i];
  float scale = proj ? (1.0f / NSEQ) : 1.0f;
  _Float16* dst = proj ? rtT : ltT;
  int rowbase = blockIdx.x * 16;
#pragma unroll
  for (int r = 0; r < 8; ++r) {
    int M = r + ((lane >> 4) << 3);
    int gr = rowbase + M;           // gr = s*L + l
    int sidx = gr >> 8, lidx = gr & 255;
    float val = (acc[r] + bv) * scale;
    dst[((long)lidx * DH + i) * NSEQ + sidx] = (_Float16)val;
  }
}

// ------------------------------------------------------------------
// 2) Fused outer-product GEMM (8192x8192x128) + proj_out (K=1024)
//    Each block: 128x128 tile of op (16 (l,m) pairs) -> out[lm][128]
// ------------------------------------------------------------------
__global__ __launch_bounds__(256) void k_outer_proj(
    const _Float16* __restrict__ ltT, const _Float16* __restrict__ rtT,
    const _Float16* __restrict__ WoT, const float* __restrict__ bo,
    _Float16* __restrict__ cat) {
  __shared__ _Float16 smA[128 * 128];
  __shared__ _Float16 smB[128 * 128];
  int tid = threadIdx.x, lane = tid & 31, w = tid >> 5;
  int bm = blockIdx.x & 63, bn = blockIdx.x >> 6;
  {
    int r = tid >> 1, h = tid & 1;
    const v4u* ga = (const v4u*)(ltT + (long)(bm * 128 + r) * 128 + h * 64);
    const v4u* gb = (const v4u*)(rtT + (long)(bn * 128 + r) * 128 + h * 64);
    v4u* la = (v4u*)(smA + r * 128 + h * 64);
    v4u* lb = (v4u*)(smB + r * 128 + h * 64);
#pragma unroll
    for (int i = 0; i < 8; ++i) { la[i] = ga[i]; lb[i] = gb[i]; }
  }
  __syncthreads();

  int wm = w & 3, wn = w >> 2;     // 4x2 wave grid: 32 M-rows x 64 N-cols each
  v8f acc[2][4];
#pragma unroll
  for (int mt = 0; mt < 2; ++mt)
#pragma unroll
    for (int nt = 0; nt < 4; ++nt) acc[mt][nt] = vzero8();
#pragma unroll
  for (int kk = 0; kk < 4; ++kk) {
    v16h a0 = load_frag(smA + (wm * 32) * 128 + kk * 32, 128, lane);
    v16h a1 = load_frag(smA + (wm * 32 + 16) * 128 + kk * 32, 128, lane);
#pragma unroll
    for (int nt = 0; nt < 4; ++nt) {
      v16h bf = load_frag(smB + (wn * 64 + nt * 16) * 128 + kk * 32, 128, lane);
      acc[0][nt] = wmma_f16(a0, bf, acc[0][nt]);
      acc[1][nt] = wmma_f16(a1, bf, acc[1][nt]);
    }
  }
  __syncthreads();

  // restage op tile as A2: [pair(16)][k2=i*32+j (1024)] f16 in LDS (reuse smA)
  _Float16* A2 = smA;
#pragma unroll
  for (int mt = 0; mt < 2; ++mt)
#pragma unroll
    for (int nt = 0; nt < 4; ++nt)
#pragma unroll
      for (int r = 0; r < 8; ++r) {
        int i = mt * 16 + r + ((lane >> 4) << 3);
        int col = wn * 64 + nt * 16 + (lane & 15);
        int pm = col >> 5, j = col & 31;
        int pr = (wm << 2) | pm;
        A2[pr * 1024 + i * 32 + j] = (_Float16)acc[mt][nt][r];
      }
  __syncthreads();

  // second GEMM: [16 pairs] x [K=1024] x [N=128]; wave w -> c-block w
  v8f acc2 = vzero8();
#pragma unroll 4
  for (int kk = 0; kk < 32; ++kk) {
    v16h a = load_frag(A2 + kk * 32, 1024, lane);
    v16h bf = load_frag(WoT + (w * 16) * 1024 + kk * 32, 1024, lane);
    acc2 = wmma_f16(a, bf, acc2);
  }
  int c = w * 16 + (lane & 15);
  float bv = bo[c];
#pragma unroll
  for (int r = 0; r < 8; ++r) {
    int pr = r + ((lane >> 4) << 3);
    int pl = pr >> 2, pm = pr & 3;
    long l = 4 * bm + pl, m = 4 * bn + pm;
    long lm = (l << 8) + m;
    cat[lm * 256 + 128 + c] = (_Float16)(acc2[r] + bv);
  }
}

// ------------------------------------------------------------------
// 3) proj_down: [65536 x 256] @ [256 x 128] -> padded NHWC f16 p_pad
// ------------------------------------------------------------------
__global__ __launch_bounds__(128) void k_proj_down(
    const _Float16* __restrict__ cat, const _Float16* __restrict__ WdT,
    const float* __restrict__ bd, _Float16* __restrict__ p_pad) {
  int tid = threadIdx.x, lane = tid & 31, w = tid >> 5;
  long r0 = (long)blockIdx.x * 64 + w * 16;
  v8f acc[8];
#pragma unroll
  for (int nt = 0; nt < 8; ++nt) acc[nt] = vzero8();
#pragma unroll
  for (int kk = 0; kk < 8; ++kk) {
    v16h a = load_frag(cat + r0 * 256 + kk * 32, 256, lane);
#pragma unroll
    for (int nt = 0; nt < 8; ++nt) {
      v16h bf = load_frag(WdT + nt * 16 * 256 + kk * 32, 256, lane);
      acc[nt] = wmma_f16(a, bf, acc[nt]);
    }
  }
#pragma unroll
  for (int nt = 0; nt < 8; ++nt) {
    int c = nt * 16 + (lane & 15);
    float bv = bd[c];
#pragma unroll
    for (int r = 0; r < 8; ++r) {
      long row = r0 + r + ((lane >> 4) << 3);
      int l = (int)(row >> 8), m = (int)(row & 255);
      p_pad[((long)(l + 1) * PAD + (m + 1)) * DP + c] = (_Float16)(acc[nt][r] + bv);
    }
  }
}

// ------------------------------------------------------------------
// 4) 3x3 conv as implicit GEMM over 9 taps.
//    Block = 64 consecutive x (fixed y). The 3x66x128 f16 input halo
//    (~50 KB) is staged once in LDS; all A fragments then come from
//    ds_load_b128 instead of 9x redundant global traffic.
// ------------------------------------------------------------------
__global__ __launch_bounds__(128) void k_conv3x3(
    const _Float16* __restrict__ in_pad, const _Float16* __restrict__ wt,
    float* __restrict__ out) {
  __shared__ _Float16 smIn[3 * 66 * DP];   // [dy+1][px-x0b][c], 50688 B
  int tid = threadIdx.x, lane = tid & 31, w = tid >> 5;
  int r0 = blockIdx.x * 64;                // 64 outputs, fixed y
  int y = r0 >> 8, x0b = r0 & 255;

  // stage halo: pad rows y..y+2, pad cols x0b..x0b+65 (borders pre-zeroed)
  for (int seg = tid; seg < 3 * 66; seg += 128) {
    int rr = seg / 66, cc = seg % 66;
    const v4u* gsrc =
        (const v4u*)(in_pad + ((long)(y + rr) * PAD + (x0b + cc)) * DP);
    v4u* ldst = (v4u*)(smIn + (rr * 66 + cc) * DP);
#pragma unroll
    for (int i = 0; i < 16; ++i) ldst[i] = gsrc[i];
  }
  __syncthreads();

  int xw = w * 16;                          // this wave's x offset in block
  v8f acc[8];
#pragma unroll
  for (int nt = 0; nt < 8; ++nt) acc[nt] = vzero8();
  for (int tap = 0; tap < 9; ++tap) {
    int dy = tap / 3, dx = tap % 3;        // 0..2 (pad-relative)
    const _Float16* abase = smIn + (dy * 66 + xw + dx) * DP;
    const _Float16* wb = wt + tap * DP * DP;
#pragma unroll
    for (int kk = 0; kk < 4; ++kk) {
      v16h a = load_frag(abase + kk * 32, DP, lane);
#pragma unroll
      for (int nt = 0; nt < 8; ++nt) {
        v16h bf = load_frag(wb + nt * 16 * DP + kk * 32, DP, lane);
        acc[nt] = wmma_f16(a, bf, acc[nt]);
      }
    }
  }
#pragma unroll
  for (int nt = 0; nt < 8; ++nt) {
    int c = nt * 16 + (lane & 15);
#pragma unroll
    for (int r = 0; r < 8; ++r) {
      int row = r0 + xw + r + ((lane >> 4) << 3);
      out[(long)row * DP + c] = acc[nt][r];
    }
  }
}

// ------------------------------------------------------------------
// 5) per-channel sum / sumsq over spatial (InstanceNorm stats)
// ------------------------------------------------------------------
__global__ __launch_bounds__(128) void k_colreduce(
    const float* __restrict__ in, float* __restrict__ sum,
    float* __restrict__ sumsq) {
  int c = threadIdx.x;
  long r0 = (long)blockIdx.x * 512;
  float s = 0.f, s2 = 0.f;
  for (long r = r0; r < r0 + 512; ++r) {
    float v = in[r * DP + c];
    s += v; s2 += v * v;
  }
  atomicAdd(&sum[c], s);
  atomicAdd(&sumsq[c], s2);
}

// ------------------------------------------------------------------
// 6) apply InstanceNorm + ELU, write padded f16 (conv2 input)
// ------------------------------------------------------------------
__global__ __launch_bounds__(256) void k_in_elu_pad(
    const float* __restrict__ hin, const float* __restrict__ sum,
    const float* __restrict__ sumsq, const float* __restrict__ gg,
    const float* __restrict__ bb, _Float16* __restrict__ out_pad) {
  long idx = (long)blockIdx.x * 256 + threadIdx.x;
  int c = (int)(idx & 127);
  long row = idx >> 7;
  float mean = sum[c] * (1.0f / (float)LM);
  float var = sumsq[c] * (1.0f / (float)LM) - mean * mean;
  float rstd = rsqrtf(var + 1e-6f);
  float v = (hin[idx] - mean) * rstd * gg[c] + bb[c];
  v = v > 0.f ? v : (__expf(v) - 1.f);
  int l = (int)(row >> 8), m = (int)(row & 255);
  out_pad[((long)(l + 1) * PAD + (m + 1)) * DP + c] = (_Float16)v;
}

// ------------------------------------------------------------------
// 7) final: out = elu(p + IN2(h2)), NHWC f32
// ------------------------------------------------------------------
__global__ __launch_bounds__(256) void k_final(
    const _Float16* __restrict__ p_pad, const float* __restrict__ h2,
    const float* __restrict__ sum, const float* __restrict__ sumsq,
    const float* __restrict__ gg, const float* __restrict__ bb,
    float* __restrict__ out) {
  long idx = (long)blockIdx.x * 256 + threadIdx.x;
  int c = (int)(idx & 127);
  long row = idx >> 7;
  int l = (int)(row >> 8), m = (int)(row & 255);
  float x = (float)p_pad[((long)(l + 1) * PAD + (m + 1)) * DP + c];
  float mean = sum[c] * (1.0f / (float)LM);
  float var = sumsq[c] * (1.0f / (float)LM) - mean * mean;
  float rstd = rsqrtf(var + 1e-6f);
  float h = (h2[idx] - mean) * rstd * gg[c] + bb[c];
  float y = x + h;
  out[idx] = y > 0.f ? y : (__expf(y) - 1.f);
}

// ------------------------------------------------------------------
extern "C" void kernel_launch(void* const* d_in, const int* in_sizes, int n_in,
                              void* d_out, int out_size, void* d_ws,
                              size_t ws_size, hipStream_t stream) {
  const float* msa  = (const float*)d_in[0];
  const float* pair = (const float*)d_in[1];
  const float* ln_g = (const float*)d_in[2];
  const float* ln_b = (const float*)d_in[3];
  const float* Wl   = (const float*)d_in[4];
  const float* bl   = (const float*)d_in[5];
  const float* Wr   = (const float*)d_in[6];
  const float* br   = (const float*)d_in[7];
  const float* Wo   = (const float*)d_in[8];
  const float* bo   = (const float*)d_in[9];
  const float* Wd   = (const float*)d_in[10];
  const float* bd   = (const float*)d_in[11];
  const float* cv1  = (const float*)d_in[12];
  const float* in1g = (const float*)d_in[13];
  const float* in1b = (const float*)d_in[14];
  const float* cv2  = (const float*)d_in[15];
  const float* in2g = (const float*)d_in[16];
  const float* in2b = (const float*)d_in[17];
  float* out = (float*)d_out;

  // ---- workspace layout (256B aligned) ----
  char* base = (char*)d_ws;
  size_t off = 0;
  auto take = [&](size_t bytes) {
    char* p = base + off;
    off = (off + bytes + 255) & ~(size_t)255;
    return p;
  };
  _Float16* WlT  = (_Float16*)take((size_t)DH * DM * 2);        // 16 KB
  _Float16* WrT  = (_Float16*)take((size_t)DH * DM * 2);
  _Float16* WoT  = (_Float16*)take((size_t)DP * DH * DH * 2);   // 256 KB
  _Float16* WdT  = (_Float16*)take((size_t)DP * 2 * DP * 2);    // 64 KB
  _Float16* Wt1  = (_Float16*)take((size_t)9 * DP * DP * 2);    // 288 KB
  _Float16* Wt2  = (_Float16*)take((size_t)9 * DP * DP * 2);
  float*    stats = (float*)take(4 * DP * sizeof(float));       // sum1,sq1,sum2,sq2
  _Float16* ltT  = (_Float16*)take((size_t)LPOS * DH * NSEQ * 2);  // 2 MB
  _Float16* rtT  = (_Float16*)take((size_t)LPOS * DH * NSEQ * 2);  // 2 MB
  _Float16* cat  = (_Float16*)take((size_t)LM * 256 * 2);       // 33.5 MB
  _Float16* p_pad = (_Float16*)take((size_t)PAD * PAD * DP * 2);   // 17 MB
  _Float16* h1_pad = (_Float16*)take((size_t)PAD * PAD * DP * 2);  // 17 MB
  float* conv_out = (float*)cat;   // alias: conv f32 out reuses cat (same 33.5MB)

  float* sum1 = stats, * sq1 = stats + DP, * sum2 = stats + 2 * DP,
       * sq2 = stats + 3 * DP;

  long pad_dwords = (long)PAD * PAD * DP * 2 / 4;

  // 0) zero padded buffers + stats
  k_zero<<<2048, 256, 0, stream>>>((unsigned int*)p_pad, pad_dwords);
  k_zero<<<2048, 256, 0, stream>>>((unsigned int*)h1_pad, pad_dwords);
  k_zero<<<2, 256, 0, stream>>>((unsigned int*)stats, 4 * DP);

  // prep: weights transpose+f16, pair -> f16 into cat[:, 0:128]
  k_prep_weights<<<(9 * DP * DP + 255) / 256, 256, 0, stream>>>(
      Wl, Wr, Wo, Wd, cv1, cv2, WlT, WrT, WoT, WdT, Wt1, Wt2);
  k_prep_pair<<<(LM * DP) / 256, 256, 0, stream>>>(pair, cat);

  // 1) LayerNorm + projections
  k_ln_proj<<<ROWS / 16, 128, 0, stream>>>(msa, ln_g, ln_b, WlT, bl, WrT, br,
                                           ltT, rtT);

  // 2) outer-product GEMM + proj_out (fused) -> cat[:, 128:256]
  k_outer_proj<<<64 * 64, 256, 0, stream>>>(ltT, rtT, WoT, bo, cat);

  // 3) proj_down -> p_pad
  k_proj_down<<<LM / 64, 128, 0, stream>>>(cat, WdT, bd, p_pad);

  // 4) conv1 (overwrites cat region as f32) + IN1 stats + ELU -> h1_pad
  k_conv3x3<<<LM / 64, 128, 0, stream>>>(p_pad, Wt1, conv_out);
  k_colreduce<<<LM / 512, 128, 0, stream>>>(conv_out, sum1, sq1);
  k_in_elu_pad<<<(LM * DP) / 256, 256, 0, stream>>>(conv_out, sum1, sq1, in1g,
                                                    in1b, h1_pad);

  // 5) conv2 + IN2 stats
  k_conv3x3<<<LM / 64, 128, 0, stream>>>(h1_pad, Wt2, conv_out);
  k_colreduce<<<LM / 512, 128, 0, stream>>>(conv_out, sum2, sq2);

  // 6) final residual + IN2 + ELU -> d_out (NHWC f32)
  k_final<<<(LM * DP) / 256, 256, 0, stream>>>(p_pad, conv_out, sum2, sq2,
                                               in2g, in2b, out);
}
